// MultiHeadAttention_5334349382389
// MI455X (gfx1250) — matrix operand order repaired and checked by reading, not compiled, not run on hardware
//
#include <hip/hip_runtime.h>

#define NHEAD 8
#define DK 64
#define DV 64
#define DIN 512
#define DMODEL 512
#define BATCH 4
#define SEQ 1024
#define ROWS (BATCH * SEQ)   // 4096
#define HBN (NHEAD * BATCH)  // 32

typedef __attribute__((ext_vector_type(16))) _Float16 v16h;
typedef __attribute__((ext_vector_type(8)))  float    v8f;
typedef __attribute__((ext_vector_type(4)))  _Float16 v4h;

union FragU { uint4 u[2]; v16h h; };

// A-fragment (16x32 f16, row-major source): lane l16 = row M,
// half 0: K = {k0..k0+7, k0+16..k0+23}; half 1: K = {k0+8..k0+15, k0+24..k0+31}
static __device__ inline v16h load_a_frag(const _Float16* row_k0, int hlf) {
  FragU f;
  f.u[0] = *reinterpret_cast<const uint4*>(row_k0 + hlf * 8);
  f.u[1] = *reinterpret_cast<const uint4*>(row_k0 + 16 + hlf * 8);
  return f.h;
}
// B-fragment (32x16 f16): lane l16 = column N; half h holds K = {k0+8h..+7, k0+16+8h..+7},
// contiguous when B's columns are contiguous rows of the source.
static __device__ inline v16h load_b_frag(const _Float16* row_k0, int hlf) {
  FragU f;
  const _Float16* p = row_k0 + hlf * 8;
  f.u[0] = *reinterpret_cast<const uint4*>(p);
  f.u[1] = *reinterpret_cast<const uint4*>(p + 16);
  return f.h;
}
static __device__ inline v8f wmma16(v16h a, v16h b, v8f c) {
  return __builtin_amdgcn_wmma_f32_16x16x32_f16(false, a, false, b, (short)0, c,
                                                false, false);
}
// reductions within each 16-lane half of a wave32 (masks <= 8 never cross halves)
static __device__ inline float half16_max(float v) {
  #pragma unroll
  for (int m = 1; m <= 8; m <<= 1) v = fmaxf(v, __shfl_xor(v, m, 32));
  return v;
}
static __device__ inline float half16_add(float v) {
  #pragma unroll
  for (int m = 1; m <= 8; m <<= 1) v += __shfl_xor(v, m, 32);
  return v;
}

// ---------------- K0: f32 -> f16 convert ----------------
__global__ void cvt_f32_to_f16(const float4* __restrict__ in,
                               v4h* __restrict__ out, int n4) {
  int i = blockIdx.x * blockDim.x + threadIdx.x;
  if (i < n4) {
    float4 f = in[i];
    v4h h;
    h[0] = (_Float16)f.x; h[1] = (_Float16)f.y;
    h[2] = (_Float16)f.z; h[3] = (_Float16)f.w;
    out[i] = h;
  }
}

// ---------------- K1: projection GEMM  Y = X * W^T + b  (2x2 register blocking) ----
// transposeOut==0: out[(h*BATCH+b)*SEQ + n][dk]   (Q/K head-split)
// transposeOut==1: out[(h*BATCH+b)*DV + d][seq]   (V transposed for PV B-frags)
__global__ __launch_bounds__(128)
void proj_gemm(const _Float16* __restrict__ X, const _Float16* __restrict__ W,
               const float* __restrict__ bias, _Float16* __restrict__ out,
               int transposeOut) {
  int wave = blockIdx.x * 4 + (threadIdx.x >> 5);
  int lane = threadIdx.x & 31;
  int hlf = lane >> 4, l16 = lane & 15;
  int tR = wave >> 4;  // 128 row blocks of 32 rows
  int tC = wave & 15;  // 16 col blocks of 32 cols
  const _Float16* arow0 = X + (size_t)(tR * 32 + l16) * DIN;
  const _Float16* arow1 = arow0 + (size_t)16 * DIN;
  const _Float16* brow0 = W + (size_t)(tC * 32 + l16) * DIN;
  const _Float16* brow1 = brow0 + (size_t)16 * DIN;
  v8f acc[2][2] = {};
  #pragma unroll 2
  for (int k0 = 0; k0 < DIN; k0 += 32) {
    v16h a0 = load_a_frag(arow0 + k0, hlf);
    v16h a1 = load_a_frag(arow1 + k0, hlf);
    v16h b0 = load_b_frag(brow0 + k0, hlf);
    v16h b1 = load_b_frag(brow1 + k0, hlf);
    acc[0][0] = wmma16(a0, b0, acc[0][0]);
    acc[0][1] = wmma16(a0, b1, acc[0][1]);
    acc[1][0] = wmma16(a1, b0, acc[1][0]);
    acc[1][1] = wmma16(a1, b1, acc[1][1]);
  }
  #pragma unroll
  for (int cj = 0; cj < 2; ++cj) {
    int j = tC * 32 + cj * 16 + l16;  // output feature (fixed per lane)
    int h = j >> 6, d = j & 63;
    float bj = bias[j];
    #pragma unroll
    for (int qi = 0; qi < 2; ++qi) {
      #pragma unroll
      for (int r = 0; r < 8; ++r) {
        int R = tR * 32 + qi * 16 + r + hlf * 8;  // global row = b*SEQ + n
        int bb = R >> 10, nseq = R & (SEQ - 1);
        float val = acc[qi][cj][r] + bj;
        size_t idx = transposeOut
            ? ((size_t)((h * BATCH + bb) * DV + d) * SEQ + nseq)
            : ((size_t)((h * BATCH + bb) * SEQ + nseq) * DK + d);
        out[idx] = (_Float16)val;
      }
    }
  }
}

// ---------------- K2: flash attention (32 queries per wave) ----------------
__global__ __launch_bounds__(128)
void flash_attn(const _Float16* __restrict__ qh, const _Float16* __restrict__ kh,
                const _Float16* __restrict__ vT, const float* __restrict__ edge,
                _Float16* __restrict__ ctx) {
  __shared__ _Float16 lds[4 * 32 * 40];  // per-wave 32x32 P tile, stride 40
  int wsl = threadIdx.x >> 5;
  int wave = blockIdx.x * 4 + wsl;
  int lane = threadIdx.x & 31;
  int hlf = lane >> 4, l16 = lane & 15;
  int hb = wave >> 5;  // 0..31
  int qt = wave & 31;  // 0..31 (32-query tiles)
  _Float16* pl = lds + wsl * (32 * 40);

  v16h fq[2][2];
  #pragma unroll
  for (int qi = 0; qi < 2; ++qi) {
    const _Float16* qrow = qh + (size_t)(hb * SEQ + qt * 32 + qi * 16 + l16) * DK;
    fq[qi][0] = load_a_frag(qrow + 0, hlf);
    fq[qi][1] = load_a_frag(qrow + 32, hlf);
  }

  v8f o[2][4] = {};
  float rowMax[2][8], rowSum[2][8];
  #pragma unroll
  for (int qi = 0; qi < 2; ++qi)
    #pragma unroll
    for (int r = 0; r < 8; ++r) { rowMax[qi][r] = -1e30f; rowSum[qi][r] = 0.f; }

  const bool isEdge = (hb < 2);  // u[0], u[1] overwritten by edge_matrix

  for (int kt = 0; kt < SEQ / 16; kt += 2) {  // 32 keys per iteration
    v8f s[2][2];
    if (isEdge) {
      #pragma unroll
      for (int qi = 0; qi < 2; ++qi)
        #pragma unroll
        for (int t = 0; t < 2; ++t)
          #pragma unroll
          for (int r = 0; r < 8; ++r) {
            int m = qt * 32 + qi * 16 + r + hlf * 8;
            s[qi][t][r] = edge[(size_t)m * SEQ + (kt + t) * 16 + l16];
          }
    } else {
      const _Float16* krow0 = kh + (size_t)(hb * SEQ + kt * 16 + l16) * DK;
      const _Float16* krow1 = krow0 + 16 * DK;
      v16h kb[2][2];
      kb[0][0] = load_b_frag(krow0 + 0, hlf);
      kb[0][1] = load_b_frag(krow0 + 32, hlf);
      kb[1][0] = load_b_frag(krow1 + 0, hlf);
      kb[1][1] = load_b_frag(krow1 + 32, hlf);
      #pragma unroll
      for (int qi = 0; qi < 2; ++qi)
        #pragma unroll
        for (int t = 0; t < 2; ++t) {
          v8f c = {};
          c = wmma16(fq[qi][0], kb[t][0], c);
          c = wmma16(fq[qi][1], kb[t][1], c);
          #pragma unroll
          for (int r = 0; r < 8; ++r) s[qi][t][r] = c[r] * 0.125f;
        }
    }
    // online softmax update (rows m = r + 8*hlf within each q subtile)
    #pragma unroll
    for (int qi = 0; qi < 2; ++qi) {
      float nm[8], sc[8];
      #pragma unroll
      for (int r = 0; r < 8; ++r) {
        float m = half16_max(fmaxf(s[qi][0][r], s[qi][1][r]));
        nm[r] = fmaxf(rowMax[qi][r], m);
        sc[r] = __expf(rowMax[qi][r] - nm[r]);
      }
      #pragma unroll
      for (int r = 0; r < 8; ++r) {
        s[qi][0][r] = __expf(s[qi][0][r] - nm[r]);
        s[qi][1][r] = __expf(s[qi][1][r] - nm[r]);
        float ts = half16_add(s[qi][0][r] + s[qi][1][r]);
        rowSum[qi][r] = rowSum[qi][r] * sc[r] + ts;
        rowMax[qi][r] = nm[r];
        o[qi][0][r] *= sc[r]; o[qi][1][r] *= sc[r];
        o[qi][2][r] *= sc[r]; o[qi][3][r] *= sc[r];
      }
      // stage P (C layout) -> LDS
      #pragma unroll
      for (int r = 0; r < 8; ++r) {
        int m = qi * 16 + r + hlf * 8;
        pl[m * 40 + l16] = (_Float16)s[qi][0][r];
        pl[m * 40 + 16 + l16] = (_Float16)s[qi][1][r];
      }
    }
    asm volatile("s_wait_dscnt 0x0" ::: "memory");  // cross-lane LDS RAW in-wave
    // V fragments loaded once, reused for both q subtiles
    v16h vb[4];
    const _Float16* vbase = vT + (size_t)(hb * DV + l16) * SEQ + kt * 16;
    #pragma unroll
    for (int d = 0; d < 4; ++d)
      vb[d] = load_b_frag(vbase + (size_t)d * 16 * SEQ, hlf);
    v16h fp0 = load_a_frag(pl + l16 * 40, hlf);
    v16h fp1 = load_a_frag(pl + (16 + l16) * 40, hlf);
    #pragma unroll
    for (int d = 0; d < 4; ++d) o[0][d] = wmma16(fp0, vb[d], o[0][d]);
    #pragma unroll
    for (int d = 0; d < 4; ++d) o[1][d] = wmma16(fp1, vb[d], o[1][d]);
  }
  // finalize: divide by row sums, store merged-head ctx [b*SEQ+n][h*64+d]
  int h = hb >> 2, bb = hb & 3;  // hb = h*BATCH + b
  #pragma unroll
  for (int qi = 0; qi < 2; ++qi)
    #pragma unroll
    for (int r = 0; r < 8; ++r) {
      float inv = 1.0f / rowSum[qi][r];
      int m = qt * 32 + qi * 16 + r + hlf * 8;
      size_t base = (size_t)(bb * SEQ + m) * (NHEAD * DV) + h * DV;
      ctx[base + 0 * 16 + l16] = (_Float16)(o[qi][0][r] * inv);
      ctx[base + 1 * 16 + l16] = (_Float16)(o[qi][1][r] * inv);
      ctx[base + 2 * 16 + l16] = (_Float16)(o[qi][2][r] * inv);
      ctx[base + 3 * 16 + l16] = (_Float16)(o[qi][3][r] * inv);
    }
}

// ---------------- K3: output GEMM  Z = ctx * Wo^T + bo (f32 out, 2x2 blocked) ----
__global__ __launch_bounds__(128)
void out_gemm(const _Float16* __restrict__ X, const _Float16* __restrict__ W,
              const float* __restrict__ bias, float* __restrict__ out) {
  int wave = blockIdx.x * 4 + (threadIdx.x >> 5);
  int lane = threadIdx.x & 31;
  int hlf = lane >> 4, l16 = lane & 15;
  int tR = wave >> 4;
  int tC = wave & 15;
  const _Float16* arow0 = X + (size_t)(tR * 32 + l16) * DMODEL;
  const _Float16* arow1 = arow0 + (size_t)16 * DMODEL;
  const _Float16* brow0 = W + (size_t)(tC * 32 + l16) * DMODEL;
  const _Float16* brow1 = brow0 + (size_t)16 * DMODEL;
  v8f acc[2][2] = {};
  #pragma unroll 2
  for (int k0 = 0; k0 < DMODEL; k0 += 32) {
    v16h a0 = load_a_frag(arow0 + k0, hlf);
    v16h a1 = load_a_frag(arow1 + k0, hlf);
    v16h b0 = load_b_frag(brow0 + k0, hlf);
    v16h b1 = load_b_frag(brow1 + k0, hlf);
    acc[0][0] = wmma16(a0, b0, acc[0][0]);
    acc[0][1] = wmma16(a0, b1, acc[0][1]);
    acc[1][0] = wmma16(a1, b0, acc[1][0]);
    acc[1][1] = wmma16(a1, b1, acc[1][1]);
  }
  #pragma unroll
  for (int cj = 0; cj < 2; ++cj) {
    int j = tC * 32 + cj * 16 + l16;
    float bj = bias[j];
    #pragma unroll
    for (int qi = 0; qi < 2; ++qi) {
      #pragma unroll
      for (int r = 0; r < 8; ++r) {
        int R = tR * 32 + qi * 16 + r + hlf * 8;
        out[(size_t)R * DMODEL + j] = acc[qi][cj][r] + bj;
      }
    }
  }
}

extern "C" void kernel_launch(void* const* d_in, const int* in_sizes, int n_in,
                              void* d_out, int out_size, void* d_ws, size_t ws_size,
                              hipStream_t stream) {
  (void)in_sizes; (void)n_in; (void)out_size; (void)ws_size;
  const float* q    = (const float*)d_in[0];
  const float* k    = (const float*)d_in[1];
  const float* v    = (const float*)d_in[2];
  const float* edge = (const float*)d_in[3];
  const float* Wq   = (const float*)d_in[4];
  const float* bq   = (const float*)d_in[5];
  const float* Wk   = (const float*)d_in[6];
  const float* bk   = (const float*)d_in[7];
  const float* Wv   = (const float*)d_in[8];
  const float* bv   = (const float*)d_in[9];
  const float* Wo   = (const float*)d_in[10];
  const float* bo   = (const float*)d_in[11];

  char* ws = (char*)d_ws;
  size_t off = 0;
  auto carve = [&](size_t bytes) {
    char* p = ws + off;
    off += (bytes + 255) & ~(size_t)255;
    return p;
  };
  _Float16* xq  = (_Float16*)carve((size_t)ROWS * DIN * 2);     // 4 MB
  _Float16* xk  = (_Float16*)carve((size_t)ROWS * DIN * 2);
  _Float16* xv  = (_Float16*)carve((size_t)ROWS * DIN * 2);
  _Float16* wqh = (_Float16*)carve((size_t)DMODEL * DIN * 2);   // 512 KB
  _Float16* wkh = (_Float16*)carve((size_t)DMODEL * DIN * 2);
  _Float16* wvh = (_Float16*)carve((size_t)DMODEL * DIN * 2);
  _Float16* woh = (_Float16*)carve((size_t)DMODEL * DMODEL * 2);
  _Float16* qhb = (_Float16*)carve((size_t)HBN * SEQ * DK * 2); // 4 MB
  _Float16* khb = (_Float16*)carve((size_t)HBN * SEQ * DK * 2);
  _Float16* vtb = (_Float16*)carve((size_t)HBN * DV * SEQ * 2);
  _Float16* ctx = (_Float16*)carve((size_t)ROWS * NHEAD * DV * 2);

  auto cvt = [&](const float* src, _Float16* dst, int n) {
    int n4 = n / 4;
    cvt_f32_to_f16<<<dim3((n4 + 255) / 256), dim3(256), 0, stream>>>(
        (const float4*)src, (v4h*)dst, n4);
  };
  cvt(q,  xq,  ROWS * DIN);
  cvt(k,  xk,  ROWS * DIN);
  cvt(v,  xv,  ROWS * DIN);
  cvt(Wq, wqh, DMODEL * DIN);
  cvt(Wk, wkh, DMODEL * DIN);
  cvt(Wv, wvh, DMODEL * DIN);
  cvt(Wo, woh, DMODEL * DMODEL);

  dim3 gGemm(512), blk(128);  // 2048 waves, one 32x32 tile each
  proj_gemm<<<gGemm, blk, 0, stream>>>(xq, wqh, bq, qhb, 0);
  proj_gemm<<<gGemm, blk, 0, stream>>>(xk, wkh, bk, khb, 0);
  proj_gemm<<<gGemm, blk, 0, stream>>>(xv, wvh, bv, vtb, 1);

  flash_attn<<<dim3(256), blk, 0, stream>>>(qhb, khb, vtb, edge, ctx);

  out_gemm<<<gGemm, blk, 0, stream>>>(ctx, woh, bo, (float*)d_out);
}
